// RWKV_TimeMix_4217657884742
// MI455X (gfx1250) — matrix operand order repaired and checked by reading, not compiled, not run on hardware
//
#include <hip/hip_runtime.h>
#include <hip/hip_bf16.h>
#include <math.h>

// ---------------------------------------------------------------------------
// RWKV TimeMix for MI455X (gfx1250): bf16 WMMA GEMMs with async global->LDS
// double-buffered staging + 3-phase chunked wkv scan.
// ---------------------------------------------------------------------------

typedef __bf16 bf16_t;
typedef __attribute__((ext_vector_type(16))) __bf16 v16bf;
typedef __attribute__((ext_vector_type(8)))  __bf16 v8bf;
typedef __attribute__((ext_vector_type(8)))  float  v8f;

#define B_DIM 8
#define T_DIM 1024
#define C_DIM 1024
#define M_TOT (B_DIM * T_DIM)          // 8192 rows
#define K_CLAMP 60.0f

// GEMM tiling
#define BM 128
#define BN 128
#define BK 32
#define LDSS 40                        // padded LDS row stride in halves (80B)

// wkv chunked scan
#define CH 16                          // chunks along T
#define CL (T_DIM / CH)                // 64 steps per chunk

// ---------------------------------------------------------------------------
// Kernel 1: fp32 weight -> bf16
// ---------------------------------------------------------------------------
__global__ __launch_bounds__(256) void cvt_w_bf16(const float* __restrict__ w,
                                                  bf16_t* __restrict__ w16, int n) {
  int i = blockIdx.x * blockDim.x + threadIdx.x;
  if (i < n) w16[i] = (bf16_t)w[i];
}

// ---------------------------------------------------------------------------
// Kernel 2: time-shift mixing, emit bf16 xm
// ---------------------------------------------------------------------------
__global__ __launch_bounds__(256) void prep_xm(const float* __restrict__ x,
                                               const float* __restrict__ time_mix,
                                               const float* __restrict__ combined_mix,
                                               bf16_t* __restrict__ xm16) {
  int idx = blockIdx.x * blockDim.x + threadIdx.x;   // over B*T*C
  int c = idx & (C_DIM - 1);
  int t = (idx >> 10) & (T_DIM - 1);
  float xv  = x[idx];
  float xm1 = (t > 0)         ? x[idx - C_DIM] : 0.0f;
  float xp1 = (t < T_DIM - 1) ? x[idx + C_DIM] : 0.0f;
  float comb = (c < (C_DIM / 2)) ? xm1 : xp1;
  float tm = time_mix[c];
  float cm = combined_mix[c];
  float v = xv * tm + xm1 * (1.0f - tm) + comb * cm;
  xm16[idx] = (bf16_t)v;
}

// ---------------------------------------------------------------------------
// Kernel 3: bf16 GEMM  Cout[m,n] = sum_k A[m,k] * W[n,k]   (einsum btc,dc->btd)
// Double-buffered LDS, staged with global_load_async_to_lds_b128 (ASYNCcnt).
// 256 threads = 8 waves; wave grid 4(M)x2(N); each wave -> 32x64 output.
// mode 0: plain f32 store; mode 1: exp(min(acc, 60))  (k projection)
// ---------------------------------------------------------------------------
__global__ __launch_bounds__(256) void gemm_bf16(const bf16_t* __restrict__ A,
                                                 const bf16_t* __restrict__ Bw,
                                                 float* __restrict__ Cout,
                                                 int M, int N, int K, int mode) {
  __shared__ __align__(16) bf16_t As[2][BM * LDSS];
  __shared__ __align__(16) bf16_t Bs[2][BN * LDSS];

  const int tid  = threadIdx.x;
  const int lane = tid & 31;
  const int wave = tid >> 5;         // 0..7
  const int wm   = wave >> 1;        // 0..3 : 32-row stripe
  const int wn   = wave & 1;         // 0..1 : 64-col stripe
  const int half = lane >> 4;        // ISA lane-half
  const int lrow = lane & 15;

  const int m0 = blockIdx.y * BM;
  const int n0 = blockIdx.x * BN;

  v8f acc[2][4];
#pragma unroll
  for (int mi = 0; mi < 2; ++mi)
#pragma unroll
    for (int ni = 0; ni < 4; ++ni) acc[mi][ni] = (v8f){0,0,0,0,0,0,0,0};

  // Each thread issues 4 async b128 copies per K-tile (2 for A, 2 for B).
  auto issue_tile = [&](int k0, int buf) {
#pragma unroll
    for (int i = 0; i < 2; ++i) {
      int chunk = tid * 2 + i;
      int r  = chunk >> 2;
      int cc = (chunk & 3) * 8;
      unsigned la = (unsigned)(uintptr_t)&As[buf][r * LDSS + cc];  // addr[31:0] = LDS offset
      const bf16_t* ga = &A[(size_t)(m0 + r) * K + k0 + cc];
      asm volatile("global_load_async_to_lds_b128 %0, %1, off"
                   :: "v"(la), "v"(ga) : "memory");
      unsigned lb = (unsigned)(uintptr_t)&Bs[buf][r * LDSS + cc];
      const bf16_t* gb = &Bw[(size_t)(n0 + r) * K + k0 + cc];
      asm volatile("global_load_async_to_lds_b128 %0, %1, off"
                   :: "v"(lb), "v"(gb) : "memory");
    }
  };

  const int NT = K / BK;
  issue_tile(0, 0);                       // prologue: tile 0 -> buffer 0

  for (int it = 0; it < NT; ++it) {
    const int cur = it & 1;
    const bool more = (it + 1 < NT);
    if (more) {
      issue_tile((it + 1) * BK, cur ^ 1); // overlap: next tile -> other buffer
      asm volatile("s_wait_asynccnt 0x4" ::: "memory");  // tile `it` landed
    } else {
      asm volatile("s_wait_asynccnt 0x0" ::: "memory");
    }
    __syncthreads();

#pragma unroll
    for (int mi = 0; mi < 2; ++mi) {
      // A fragment (16x32 bf16, ISA layout): lane half selects K-subsets
      int arow = wm * 32 + mi * 16 + lrow;
      v8bf alo = *(const v8bf*)&As[cur][arow * LDSS + half * 8];      // K h*8..+7
      v8bf ahi = *(const v8bf*)&As[cur][arow * LDSS + 16 + half * 8]; // K 16+h*8..+7
      v16bf afrag = __builtin_shufflevector(alo, ahi,
          0,1,2,3,4,5,6,7,8,9,10,11,12,13,14,15);
#pragma unroll
      for (int ni = 0; ni < 4; ++ni) {
        // B fragment (32x16 bf16): lane = column, half selects K h*8..+7 and 16+h*8..+7
        int bcol = wn * 64 + ni * 16 + lrow;
        v8bf b0 = *(const v8bf*)&Bs[cur][bcol * LDSS + half * 8];
        v8bf b1 = *(const v8bf*)&Bs[cur][bcol * LDSS + 16 + half * 8];
        v16bf bfrag = __builtin_shufflevector(b0, b1,
            0,1,2,3,4,5,6,7,8,9,10,11,12,13,14,15);
        acc[mi][ni] = __builtin_amdgcn_wmma_f32_16x16x32_bf16(
            false, afrag, false, bfrag, (short)0, acc[mi][ni], false, false);
      }
    }
    __syncthreads();   // all waves done reading buf[cur] before it is reused
  }

  // ---- epilogue (C/D layout: vgpr i -> row i + half*8, col = lane&15)
#pragma unroll
  for (int mi = 0; mi < 2; ++mi) {
#pragma unroll
    for (int ni = 0; ni < 4; ++ni) {
      int rbase = m0 + wm * 32 + mi * 16 + half * 8;
      int col   = n0 + wn * 64 + ni * 16 + lrow;
#pragma unroll
      for (int i = 0; i < 8; ++i) {
        float v = acc[mi][ni][i];
        if (mode == 1) v = __expf(fminf(v, K_CLAMP));
        Cout[(size_t)(rbase + i) * N + col] = v;
      }
    }
  }
}

// ---------------------------------------------------------------------------
// wkv chunked scan. State (a,b,p) composes associatively:
//   merge((a1,b1,p1),(a2,b2,p2)): pn=max(p1,p2);
//   (e^{p1-pn}a1 + e^{p2-pn}a2, e^{p1-pn}b1 + e^{p2-pn}b2, pn)
// One step = decay incoming by w, merge with (v, 1, k).
// ---------------------------------------------------------------------------

// Phase 1: per-(b,c,chunk) aggregate from identity. g = (b*CH + j)*C + c.
__global__ __launch_bounds__(256) void wkv_chunk(const float* __restrict__ kf,
                                                 const float* __restrict__ vf,
                                                 const float* __restrict__ time_decay,
                                                 float* __restrict__ CA,
                                                 float* __restrict__ CB,
                                                 float* __restrict__ CP) {
  int g = blockIdx.x * blockDim.x + threadIdx.x;   // B*CH*C threads
  int c = g & (C_DIM - 1);
  int j = (g >> 10) & (CH - 1);
  int b = g >> 14;
  const float w = time_decay[c] * (1.0f / (float)T_DIM);

  float a = 0.0f, bb = 0.0f, p = -1e38f;
  size_t base = ((size_t)b * T_DIM + j * CL) * C_DIM + c;
  for (int t = 0; t < CL; ++t) {
    size_t off = base + (size_t)t * C_DIM;
    float kt = kf[off];
    float vt = vf[off];
    float no2 = fmaxf(w + p, kt);
    float f1  = __expf(w + p - no2);
    float f2  = __expf(kt - no2);
    a  = f1 * a + f2 * vt;
    bb = f1 * bb + f2;
    p  = no2;
  }
  CA[g] = a; CB[g] = bb; CP[g] = p;
}

// Phase 2: fold chunk aggregates into per-chunk incoming states, in place.
__global__ __launch_bounds__(256) void wkv_prefix(const float* __restrict__ time_decay,
                                                  float* __restrict__ CA,
                                                  float* __restrict__ CB,
                                                  float* __restrict__ CP) {
  int g = blockIdx.x * blockDim.x + threadIdx.x;   // B*C threads
  int c = g & (C_DIM - 1);
  int b = g >> 10;
  const float wL = time_decay[c] * (1.0f / (float)T_DIM) * (float)CL;

  float a = 0.0f, bb = 0.0f, p = -1e38f;
  for (int j = 0; j < CH; ++j) {
    size_t idx = ((size_t)b * CH + j) * C_DIM + c;
    float ca = CA[idx], cb = CB[idx], pc = CP[idx];
    CA[idx] = a; CB[idx] = bb; CP[idx] = p;        // incoming state for chunk j
    float p1 = p + wL;                              // incoming decayed over chunk
    float pn = fmaxf(p1, pc);
    float e1 = __expf(p1 - pn);
    float e2 = __expf(pc - pn);
    a  = e1 * a  + e2 * ca;
    bb = e1 * bb + e2 * cb;
    p  = pn;
  }
}

// Phase 3: replay chunk from incoming state; emit bf16 sigmoid(r)*y.
__global__ __launch_bounds__(256) void wkv_emit(const float* __restrict__ kf,
                                                const float* __restrict__ vf,
                                                const float* __restrict__ rf,
                                                const float* __restrict__ time_decay,
                                                const float* __restrict__ time_first,
                                                const float* __restrict__ CA,
                                                const float* __restrict__ CB,
                                                const float* __restrict__ CP,
                                                bf16_t* __restrict__ a16) {
  int g = blockIdx.x * blockDim.x + threadIdx.x;   // B*CH*C threads
  int c = g & (C_DIM - 1);
  int j = (g >> 10) & (CH - 1);
  int b = g >> 14;
  const float w = time_decay[c] * (1.0f / (float)T_DIM);
  const float u = time_first[c] * (1.0f / (float)T_DIM);

  float a = CA[g], bb = CB[g], p = CP[g];
  size_t base = ((size_t)b * T_DIM + j * CL) * C_DIM + c;
  for (int t = 0; t < CL; ++t) {
    size_t off = base + (size_t)t * C_DIM;
    float kt = kf[off];
    float vt = vf[off];
    float rt = rf[off];
    // output at time t (u-boosted merge of state-before-t with (v,1,k))
    float no = fmaxf(p, u + kt);
    float e1 = __expf(p - no);
    float e2 = __expf(u + kt - no);
    float y  = (e1 * a + e2 * vt) / (e1 * bb + e2);
    // state update
    float no2 = fmaxf(w + p, kt);
    float f1  = __expf(w + p - no2);
    float f2  = __expf(kt - no2);
    a  = f1 * a + f2 * vt;
    bb = f1 * bb + f2;
    p  = no2;
    // fused sigmoid(r) * y -> bf16 for final GEMM
    float sr = 1.0f / (1.0f + __expf(-rt));
    a16[off] = (bf16_t)(sr * y);
  }
}

// ---------------------------------------------------------------------------
// Launch
// ---------------------------------------------------------------------------
extern "C" void kernel_launch(void* const* d_in, const int* in_sizes, int n_in,
                              void* d_out, int out_size, void* d_ws, size_t ws_size,
                              hipStream_t stream) {
  const float* x            = (const float*)d_in[0];
  const float* time_decay   = (const float*)d_in[1];
  const float* time_first   = (const float*)d_in[2];
  const float* time_mix     = (const float*)d_in[3];
  const float* combined_mix = (const float*)d_in[4];
  const float* Wk           = (const float*)d_in[5];
  const float* Wv           = (const float*)d_in[6];
  const float* Wr           = (const float*)d_in[7];
  const float* Wo           = (const float*)d_in[8];
  float* out = (float*)d_out;

  char* ws = (char*)d_ws;
  const size_t MB = 1u << 20;
  bf16_t* xm16 = (bf16_t*)(ws);             // 16 MB; reused as a16 after GEMMs
  bf16_t* Wk16 = (bf16_t*)(ws + 16 * MB);   // 2 MB each
  bf16_t* Wv16 = (bf16_t*)(ws + 18 * MB);
  bf16_t* Wr16 = (bf16_t*)(ws + 20 * MB);
  bf16_t* Wo16 = (bf16_t*)(ws + 22 * MB);
  float*  kf   = (float*)(ws + 24 * MB);    // 32 MB each
  float*  vf   = (float*)(ws + 56 * MB);
  float*  rf   = (float*)(ws + 88 * MB);
  float*  CA   = (float*)(ws + 120 * MB);   // B*CH*C floats = 0.5 MB each
  float*  CB   = (float*)(ws + 122 * MB);
  float*  CP   = (float*)(ws + 124 * MB);

  const int CW = C_DIM * C_DIM;             // weight elements
  cvt_w_bf16<<<CW / 256, 256, 0, stream>>>(Wk, Wk16, CW);
  cvt_w_bf16<<<CW / 256, 256, 0, stream>>>(Wv, Wv16, CW);
  cvt_w_bf16<<<CW / 256, 256, 0, stream>>>(Wr, Wr16, CW);
  cvt_w_bf16<<<CW / 256, 256, 0, stream>>>(Wo, Wo16, CW);

  prep_xm<<<(M_TOT * C_DIM) / 256, 256, 0, stream>>>(x, time_mix, combined_mix, xm16);

  dim3 gg(C_DIM / BN, M_TOT / BM);          // (8, 64)
  gemm_bf16<<<gg, 256, 0, stream>>>(xm16, Wk16, kf, M_TOT, C_DIM, C_DIM, 1);
  gemm_bf16<<<gg, 256, 0, stream>>>(xm16, Wv16, vf, M_TOT, C_DIM, C_DIM, 0);
  gemm_bf16<<<gg, 256, 0, stream>>>(xm16, Wr16, rf, M_TOT, C_DIM, C_DIM, 0);

  // Chunked wkv scan (16x parallelism on the latency-bound recurrence)
  bf16_t* a16 = xm16;                       // reuse xm16 buffer
  const int NCHUNK = B_DIM * CH * C_DIM;    // 131072
  wkv_chunk<<<NCHUNK / 256, 256, 0, stream>>>(kf, vf, time_decay, CA, CB, CP);
  wkv_prefix<<<(B_DIM * C_DIM) / 256, 256, 0, stream>>>(time_decay, CA, CB, CP);
  wkv_emit<<<NCHUNK / 256, 256, 0, stream>>>(kf, vf, rf, time_decay, time_first,
                                             CA, CB, CP, a16);

  gemm_bf16<<<gg, 256, 0, stream>>>(a16, Wo16, out, M_TOT, C_DIM, C_DIM, 0);
}
